// MultiheadAttention_56831007260786
// MI455X (gfx1250) — matrix operand order repaired and checked by reading, not compiled, not run on hardware
//
#include <hip/hip_runtime.h>

// ---------------- problem constants (match reference) ----------------
constexpr int kB      = 2;
constexpr int kS      = 4096;
constexpr int kDin    = 512;
constexpr int kE      = 512;
constexpr int kH      = 8;
constexpr int kHD     = 64;              // head dim
constexpr int kWin    = 128;
constexpr int kChunks = kS / kWin;       // 32
constexpr int kM      = kB * kS;         // 8192 GEMM rows
constexpr int k3E     = 3 * kE;          // 1536
constexpr float kScale = 0.125f;         // 1/sqrt(64)
constexpr float kNeg   = -1e30f;

typedef _Float16 h8  __attribute__((ext_vector_type(8)));
typedef _Float16 h16 __attribute__((ext_vector_type(16)));
typedef float    f8  __attribute__((ext_vector_type(8)));

typedef unsigned int u32x4 __attribute__((ext_vector_type(4)));
typedef int          i32x4 __attribute__((ext_vector_type(4)));
typedef int          i32x8 __attribute__((ext_vector_type(8)));

#if __has_builtin(__builtin_amdgcn_tensor_load_to_lds) && \
    __has_builtin(__builtin_amdgcn_s_wait_tensorcnt)
#define USE_TDM 1
#else
#define USE_TDM 0
#endif

__device__ __forceinline__ f8 zero_f8() {
  f8 z = {0.f, 0.f, 0.f, 0.f, 0.f, 0.f, 0.f, 0.f};
  return z;
}

#if USE_TDM
// Issue a TDM 2D f16 tile load: tile (tile_x elems) x (tile_y rows) from a
// row-major tensor with row stride `stride_x` elems, into LDS at lds_byte_off
// (rows stored contiguously, tile_x elems each). D# packing per ISA 08 §8.3-8.5.
// This toolchain exposes the 6-arg builtin: (g0, g1, g2, g3, g4, cpol).
__device__ __forceinline__ void tdm_load_2d_f16(const _Float16* gsrc,
                                                unsigned lds_byte_off,
                                                int tile_x, int tile_y,
                                                int stride_x,
                                                int tensor_x, int tensor_y) {
  unsigned long long ga = (unsigned long long)gsrc;
  u32x4 g0;
  g0.x = 1u;                                             // count=1, user D#
  g0.y = lds_byte_off;                                   // lds_addr
  g0.z = (unsigned)(ga & 0xFFFFFFFFu);                   // global_addr[31:0]
  g0.w = (unsigned)((ga >> 32) & 0x01FFFFFFu) | (2u << 30); // [56:32] | type=2
  i32x8 g1;
  g1[0] = 1 << 16;                                       // data_size=1 (2B)
  g1[1] = (tensor_x & 0xFFFF) << 16;                     // tensor_dim0[15:0]
  g1[2] = ((tensor_x >> 16) & 0xFFFF) | ((tensor_y & 0xFFFF) << 16);
  g1[3] = ((tensor_y >> 16) & 0xFFFF) | ((tile_x & 0xFFFF) << 16);
  g1[4] = tile_y & 0xFFFF;                               // tile_dim1, tile_dim2=0
  g1[5] = stride_x;                                      // tensor_dim0_stride
  g1[6] = 0;
  g1[7] = 0;
  i32x4 z4 = {0, 0, 0, 0};
  i32x8 z8 = {0, 0, 0, 0, 0, 0, 0, 0};
  __builtin_amdgcn_tensor_load_to_lds(g0, g1, z4, z4, z8, 0);
}
__device__ __forceinline__ unsigned lds_off(const void* p) {
  return (unsigned)(unsigned long long)p;                // LDS offset = low 32b
}
#endif

// A-matrix 16x32 f16 fragment (wave32). Row-major source, `stride` halves/row.
// Lane layout: lane<16 -> M=lane, K = {0..15};
//              lane>=16 -> M=lane-16, K = {16..31}.
__device__ __forceinline__ h16 load_a_frag(const _Float16* base, int stride) {
  const int lane = threadIdx.x & 31;
  const int m    = lane & 15;
  const int kb   = (lane < 16) ? 0 : 16;
  const _Float16* p = base + m * stride + kb;
  return *(const h16*)(p);
}

// B-matrix 32x16 f16 fragment. Source laid out so the 32 K-values of output
// column n are contiguous: base[n*stride + k].
__device__ __forceinline__ h16 load_b_frag(const _Float16* base, int stride) {
  const int lane = threadIdx.x & 31;
  const int n    = lane & 15;
  const int kb   = (lane < 16) ? 0 : 16;
  return *(const h16*)(base + n * stride + kb);
}

__device__ __forceinline__ f8 wmma_f32_f16(h16 a, h16 b, f8 c) {
  return __builtin_amdgcn_wmma_f32_16x16x32_f16(false, a, false, b, (short)0, c,
                                                false, false);
}

// ---------------- kernel 1: fused QKV projection ----------------
// C = x(8192x512) @ w_qkv(512x1536) + b_qkv, scattered to f16 q/k/v buffers
// laid out [b*H + h][s][hd] so attention reads rows contiguously.
__global__ __launch_bounds__(256) void qkv_kernel(
    const float* __restrict__ x, const float* __restrict__ w,
    const float* __restrict__ bias, _Float16* __restrict__ q,
    _Float16* __restrict__ k, _Float16* __restrict__ v) {
  __shared__ __align__(32) _Float16 Alds[64 * 32];   // [m][k]
  __shared__ __align__(32) _Float16 Blds[64 * 32];   // transposed: [n][k]

  const int tid  = threadIdx.x;
  const int wave = tid >> 5;
  const int lane = tid & 31;
  const int ln   = lane & 15;
  const int rb   = (lane < 16) ? 0 : 8;
  const int wm   = wave >> 1;   // 0..3 : 16-row slice
  const int wn   = wave & 1;    // 0..1 : 32-col slice

  const int m0 = blockIdx.x * 64;
  const int n0 = blockIdx.y * 64;

  f8 c0 = zero_f8(), c1 = zero_f8();

  for (int k0 = 0; k0 < kDin; k0 += 32) {
    __syncthreads();
#pragma unroll
    for (int i = 0; i < 8; ++i) {
      int idx = tid + i * 256;                       // 64x32 = 2048 elements
      int am = idx >> 5, ak = idx & 31;
      Alds[idx] = (_Float16)x[(size_t)(m0 + am) * kDin + (k0 + ak)];
      int bn = idx & 63, bk = idx >> 6;              // transpose W into [n][k]
      Blds[bn * 32 + bk] = (_Float16)w[(size_t)(k0 + bk) * k3E + (n0 + bn)];
    }
    if (k0 + 32 < kDin)
      __builtin_prefetch(&w[(size_t)(k0 + 32) * k3E + n0 + (tid & 63)], 0, 0);
    __syncthreads();

    h16 a  = load_a_frag(Alds + wm * 16 * 32, 32);
    h16 b0 = load_b_frag(Blds + (wn * 32 + 0) * 32, 32);
    h16 b1 = load_b_frag(Blds + (wn * 32 + 16) * 32, 32);
    c0 = wmma_f32_f16(a, b0, c0);
    c1 = wmma_f32_f16(a, b1, c1);
  }

  // epilogue: bias + scatter. n layout: h*192 + {0..63 q | 64..127 k | 128..191 v}
  const int which = (n0 % 192) / 64;
  const int head  = n0 / 192;
  _Float16* dst = (which == 0) ? q : (which == 1) ? k : v;

#pragma unroll
  for (int t = 0; t < 2; ++t) {
    const int n   = n0 + wn * 32 + t * 16 + ln;
    const int hd  = n & 63;
    const float bv = bias[n];
    const f8 c = t ? c1 : c0;
#pragma unroll
    for (int r = 0; r < 8; ++r) {
      const int gm = m0 + wm * 16 + rb + r;          // = b*kS + s
      const int bb = gm >> 12;                       // / kS
      const int ss = gm & (kS - 1);
      dst[(((size_t)bb * kH + head) * kS + ss) * kHD + hd] =
          (_Float16)(c[r] + bv);
    }
  }
}

// ---------------- kernel 2: fused sliding-window attention ----------------
// One workgroup per (chunk, b*H + h). 8 waves x 16 query rows = 128 queries.
// Flash-style online softmax over 12 key tiles of 32 keys (384-key window).
// Since S and window offsets are multiples of 32, every key tile is fully
// in-range or fully skipped -> TDM can DMA the K tile with no masking.
__global__ __launch_bounds__(256) void attn_kernel(
    const _Float16* __restrict__ Q, const _Float16* __restrict__ K,
    const _Float16* __restrict__ V, _Float16* __restrict__ vals) {
  __shared__ __align__(32) _Float16 Klds[32 * kHD];   // [key][dim]
  __shared__ __align__(32) _Float16 Vtlds[kHD * 32];  // [dim][key] (transposed)
  __shared__ __align__(32) _Float16 Plds[8][16 * 32]; // per-wave P re-swizzle

  const int tid  = threadIdx.x;
  const int wave = tid >> 5;
  const int lane = tid & 31;
  const int ln   = lane & 15;
  const int rb   = (lane < 16) ? 0 : 8;

  const int chunk = blockIdx.x;
  const int bh    = blockIdx.y;
  const int b     = bh / kH;
  const int h     = bh % kH;

  const _Float16* Qb = Q + (size_t)bh * kS * kHD;
  const _Float16* Kb = K + (size_t)bh * kS * kHD;
  const _Float16* Vb = V + (size_t)bh * kS * kHD;

  const int q0 = chunk * kWin + wave * 16;

  // Q fragments for this wave's 16 rows (K-dim 64 = two 32-wide steps)
  h16 qa0 = load_a_frag(Qb + (size_t)q0 * kHD, kHD);
  h16 qa1 = load_a_frag(Qb + (size_t)q0 * kHD + 32, kHD);

  f8 acc[4];
  float m_r[8], l_r[8];
#pragma unroll
  for (int t = 0; t < 4; ++t) acc[t] = zero_f8();
#pragma unroll
  for (int r = 0; r < 8; ++r) { m_r[r] = kNeg; l_r[r] = 0.f; }

  for (int kt = 0; kt < 12; ++kt) {
    const int key0 = chunk * kWin - kWin + kt * 32;
    if (key0 < 0 || key0 >= kS) continue;            // uniform: whole tiles only

    __syncthreads();
#if USE_TDM
    // DMA K tile (32 keys x 64 dims, contiguous rows) straight into LDS.
    if (wave == 0)
      tdm_load_2d_f16(Kb + (size_t)key0 * kHD, lds_off(Klds),
                      /*tile_x=*/kHD, /*tile_y=*/32, /*stride_x=*/kHD,
                      /*tensor_x=*/kHD, /*tensor_y=*/kS - key0);
#pragma unroll
    for (int i = 0; i < 8; ++i) {                    // V needs a transpose: VALU path
      int idx = tid + i * 256;
      int kk = idx >> 6;
      int d  = idx & 63;
      Vtlds[d * 32 + kk] = Vb[(size_t)(key0 + kk) * kHD + d];
    }
    if (wave == 0) __builtin_amdgcn_s_wait_tensorcnt(0);
#else
#pragma unroll
    for (int i = 0; i < 8; ++i) {
      int idx = tid + i * 256;                       // 32 keys x 64 dims
      int kk = idx >> 6;
      int d  = idx & 63;
      Klds[kk * kHD + d] = Kb[(size_t)(key0 + kk) * kHD + d];
      Vtlds[d * 32 + kk] = Vb[(size_t)(key0 + kk) * kHD + d];
    }
#endif
    __syncthreads();

    // logits = Q K^T : two 16x16 N-tiles over the 32 staged keys
    f8 c0 = zero_f8(), c1 = zero_f8();
    {
      h16 b00 = load_b_frag(Klds + 0 * 16 * kHD + 0, kHD);
      h16 b10 = load_b_frag(Klds + 1 * 16 * kHD + 0, kHD);
      c0 = wmma_f32_f16(qa0, b00, c0);
      c1 = wmma_f32_f16(qa0, b10, c1);
      h16 b01 = load_b_frag(Klds + 0 * 16 * kHD + 32, kHD);
      h16 b11 = load_b_frag(Klds + 1 * 16 * kHD + 32, kHD);
      c0 = wmma_f32_f16(qa1, b01, c0);
      c1 = wmma_f32_f16(qa1, b11, c1);
    }

    // online softmax in C-fragment layout: row r owned by one 16-lane half
    const int col0 = key0 + ln;
    const int col1 = key0 + 16 + ln;
#pragma unroll
    for (int r = 0; r < 8; ++r) {
      const int iq = q0 + rb + r;
      bool v0 = (col0 >= iq - kWin) && (col0 <= iq + kWin);
      bool v1 = (col1 >= iq - kWin) && (col1 <= iq + kWin);
      float l0 = v0 ? c0[r] * kScale : kNeg;
      float l1 = v1 ? c1[r] * kScale : kNeg;
      float mx = fmaxf(l0, l1);
#pragma unroll
      for (int off = 8; off >= 1; off >>= 1)
        mx = fmaxf(mx, __shfl_xor(mx, off, 16));
      float mnew = fmaxf(m_r[r], mx);
      float sc = __expf(m_r[r] - mnew);
      float p0 = v0 ? __expf(l0 - mnew) : 0.f;
      float p1 = v1 ? __expf(l1 - mnew) : 0.f;
      float rs = p0 + p1;
#pragma unroll
      for (int off = 8; off >= 1; off >>= 1)
        rs += __shfl_xor(rs, off, 16);
      l_r[r] = l_r[r] * sc + rs;
      m_r[r] = mnew;
#pragma unroll
      for (int t = 0; t < 4; ++t) acc[t][r] *= sc;
      // C-layout -> LDS (row-major) so it can be re-read as an A fragment
      Plds[wave][(rb + r) * 32 + ln]      = (_Float16)p0;
      Plds[wave][(rb + r) * 32 + 16 + ln] = (_Float16)p1;
    }

    // out += P (16x32) x V (32x64): 4 WMMAs over the hd tiles
    h16 pa = load_a_frag(&Plds[wave][0], 32);
#pragma unroll
    for (int t = 0; t < 4; ++t) {
      h16 vb = load_b_frag(Vtlds + t * 16 * 32, 32);
      acc[t] = wmma_f32_f16(pa, vb, acc[t]);
    }
  }

  // normalize and store as f16 into vals[b][s][h*64+hd]
#pragma unroll
  for (int r = 0; r < 8; ++r) {
    const int s   = q0 + rb + r;
    const float inv = 1.0f / l_r[r];
    _Float16* dst = vals + ((size_t)b * kS + s) * kE + h * kHD;
#pragma unroll
    for (int t = 0; t < 4; ++t)
      dst[t * 16 + ln] = (_Float16)(acc[t][r] * inv);
  }
}

// ---------------- kernel 3: output projection ----------------
// out = vals(8192x512, f16) @ w_o(512x512) + b_o, f32 output.
// The A tile (64 rows x 32 f16, row stride 512) is a strided 2D tile -> TDM.
__global__ __launch_bounds__(256) void oproj_kernel(
    const _Float16* __restrict__ vals, const float* __restrict__ w,
    const float* __restrict__ bias, float* __restrict__ out) {
  __shared__ __align__(32) _Float16 Alds[64 * 32];
  __shared__ __align__(32) _Float16 Blds[64 * 32];   // [n][k]

  const int tid  = threadIdx.x;
  const int wave = tid >> 5;
  const int lane = tid & 31;
  const int ln   = lane & 15;
  const int rb   = (lane < 16) ? 0 : 8;
  const int wm   = wave >> 1;
  const int wn   = wave & 1;

  const int m0 = blockIdx.x * 64;
  const int n0 = blockIdx.y * 64;

  f8 c0 = zero_f8(), c1 = zero_f8();

  for (int k0 = 0; k0 < kE; k0 += 32) {
    __syncthreads();
#if USE_TDM
    if (wave == 0)
      tdm_load_2d_f16(vals + (size_t)m0 * kE + k0, lds_off(Alds),
                      /*tile_x=*/32, /*tile_y=*/64, /*stride_x=*/kE,
                      /*tensor_x=*/kE - k0, /*tensor_y=*/kM - m0);
#pragma unroll
    for (int i = 0; i < 8; ++i) {
      int idx = tid + i * 256;
      int bn = idx & 63, bk = idx >> 6;
      Blds[bn * 32 + bk] = (_Float16)w[(size_t)(k0 + bk) * kE + (n0 + bn)];
    }
    if (wave == 0) __builtin_amdgcn_s_wait_tensorcnt(0);
#else
#pragma unroll
    for (int i = 0; i < 8; ++i) {
      int idx = tid + i * 256;
      int am = idx >> 5, ak = idx & 31;
      Alds[idx] = vals[(size_t)(m0 + am) * kE + (k0 + ak)];
      int bn = idx & 63, bk = idx >> 6;
      Blds[bn * 32 + bk] = (_Float16)w[(size_t)(k0 + bk) * kE + (n0 + bn)];
    }
#endif
    __syncthreads();

    h16 a  = load_a_frag(Alds + wm * 16 * 32, 32);
    h16 b0 = load_b_frag(Blds + (wn * 32 + 0) * 32, 32);
    h16 b1 = load_b_frag(Blds + (wn * 32 + 16) * 32, 32);
    c0 = wmma_f32_f16(a, b0, c0);
    c1 = wmma_f32_f16(a, b1, c1);
  }

#pragma unroll
  for (int t = 0; t < 2; ++t) {
    const int n = n0 + wn * 32 + t * 16 + ln;
    const float bv = bias[n];
    const f8 c = t ? c1 : c0;
#pragma unroll
    for (int r = 0; r < 8; ++r) {
      const int gm = m0 + wm * 16 + rb + r;
      out[(size_t)gm * kE + n] = c[r] + bv;
    }
  }
}

// ---------------- launch ----------------
extern "C" void kernel_launch(void* const* d_in, const int* in_sizes, int n_in,
                              void* d_out, int out_size, void* d_ws, size_t ws_size,
                              hipStream_t stream) {
  (void)in_sizes; (void)n_in; (void)out_size; (void)ws_size;

  const float* x     = (const float*)d_in[0];
  const float* w_qkv = (const float*)d_in[1];
  const float* b_qkv = (const float*)d_in[2];
  const float* w_o   = (const float*)d_in[3];
  const float* b_o   = (const float*)d_in[4];
  float* out = (float*)d_out;

  // workspace: q, k, v, vals as f16 (4 x 8 MB = 32 MB)
  const size_t perBuf = (size_t)kB * kH * kS * kHD;
  _Float16* q    = (_Float16*)d_ws;
  _Float16* k    = q + perBuf;
  _Float16* v    = k + perBuf;
  _Float16* vals = v + perBuf;

  qkv_kernel<<<dim3(kM / 64, k3E / 64), 256, 0, stream>>>(x, w_qkv, b_qkv, q, k, v);
  attn_kernel<<<dim3(kChunks, kB * kH), 256, 0, stream>>>(q, k, v, vals);
  oproj_kernel<<<dim3(kM / 64, kE / 64), 256, 0, stream>>>(vals, w_o, b_o, out);
}
